// Model_24730421690438
// MI455X (gfx1250) — matrix operand order repaired and checked by reading, not compiled, not run on hardware
//
#include <hip/hip_runtime.h>
#include <hip/hip_bf16.h>

// ---------------- problem constants ----------------
#define B_    8192
#define T_    128
#define Z_    128
#define ZT_   129
#define FLAT_ 16512          // T_*ZT_
#define KS_   16
#define NH_   32
#define NR_   16
#define NQ_   1024           // 2*NH*KS (Q1 heads then Q2 heads)
#define WKLD_ 160            // padded W_k K-dim (multiple of 32)

#define QG_KC 64             // K chunk staged in LDS per iteration
#define SB_LD 72             // padded LDS row stride (halves) -> conflict-free B-frag reads

typedef __attribute__((ext_vector_type(16))) _Float16 v16h;
typedef __attribute__((ext_vector_type(8)))  _Float16 v8h;
typedef __attribute__((ext_vector_type(8)))  float    v8f;
typedef __attribute__((ext_vector_type(4)))  int      v4i;

#if defined(__AMDGCN__) && __has_builtin(__builtin_amdgcn_global_load_async_to_lds_b128)
#define HAS_ASYNC_LDS 1
#else
#define HAS_ASYNC_LDS 0
#endif

// ---------------- WMMA helpers ----------------
__device__ __forceinline__ v8f wmma_f16(v16h a, v16h b, v8f c) {
  // D = A(16x32 f16) * B(32x16 f16) + C(16x16 f32)
  return __builtin_amdgcn_wmma_f32_16x16x32_f16(
      /*neg_a=*/false, a, /*neg_b=*/false, b,
      /*c_mod=*/(short)0, c, /*reuse_a=*/false, /*reuse_b=*/false);
}

__device__ __forceinline__ v16h combine16(v8h lo, v8h hi) {
  v16h a;
#pragma unroll
  for (int i = 0; i < 8; i++) { a[i] = lo[i]; a[i + 8] = hi[i]; }
  return a;
}

// A fragment: 16x32 f16 tile from row-major matrix (row stride ld halves).
// Per ISA: lanes 0-15 (M=lane) hold K {k0..k0+7, k0+16..k0+23};
//          lanes 16-31 (M=lane-16) hold K {k0+8..k0+15, k0+24..k0+31}.
__device__ __forceinline__ v16h load_a_frag(const _Float16* base, int ld,
                                            int m0, int k0, int lane) {
  int lh = lane & 15, sel = lane >> 4;
  const _Float16* p = base + (size_t)(m0 + lh) * ld + k0 + sel * 8;
  return combine16(*(const v8h*)p, *(const v8h*)(p + 16));
}

// B fragment: 32x16 f16 tile, sourced from BT[n][k] row-major (ld halves).
// Per ISA: lane n (n=lane&15) holds K {k0+8*(lane>>4)..+7, k0+16+8*(lane>>4)..+7}.
__device__ __forceinline__ v16h load_b_frag(const _Float16* bt, int ld,
                                            int n0, int k0, int lane) {
  int lh = lane & 15, sel = lane >> 4;
  const _Float16* p = bt + (size_t)(n0 + lh) * ld + k0 + sel * 8;
  return combine16(*(const v8h*)p, *(const v8h*)(p + 16));
}

// ---------------- async copy helpers (gfx1250 ASYNCcnt path) ----------------
// Builtin signature (confirmed by clang diagnostic): (v4i AS1* src,
// v4i AS3* dst, imm int offset, imm int cpol)
__device__ __forceinline__ void stage_async16(const _Float16* g, _Float16* l) {
#if HAS_ASYNC_LDS
  __builtin_amdgcn_global_load_async_to_lds_b128(
      (__attribute__((address_space(1))) v4i*)(_Float16*)g,
      (__attribute__((address_space(3))) v4i*)l, /*offset=*/0, /*cpol=*/0);
#endif
}

__device__ __forceinline__ void wait_async0() {
#if defined(__AMDGCN__) && __has_builtin(__builtin_amdgcn_s_wait_asynccnt)
  __builtin_amdgcn_s_wait_asynccnt(0);
#elif defined(__AMDGCN__)
  asm volatile("s_wait_asynccnt 0" ::: "memory");
#endif
}

// ---------------- kernel 1: weight conversion ----------------
__global__ void prep_wcat_kernel(const float* __restrict__ Wq1,
                                 const float* __restrict__ Wq2,
                                 _Float16* __restrict__ Wcat) {
  size_t idx = (size_t)blockIdx.x * 256 + threadIdx.x;
  size_t total = (size_t)NQ_ * FLAT_;
  if (idx >= total) return;
  int n = (int)(idx / FLAT_);
  int k = (int)(idx - (size_t)n * FLAT_);
  float v = (n < NH_ * KS_) ? Wq1[(size_t)n * FLAT_ + k]
                            : Wq2[(size_t)(n - NH_ * KS_) * FLAT_ + k];
  Wcat[idx] = (_Float16)v;
}

__global__ void prep_wk_kernel(const float* __restrict__ Wk,
                               _Float16* __restrict__ Wk16) {
  int idx = blockIdx.x * 256 + threadIdx.x;
  if (idx >= KS_ * WKLD_) return;
  int k = idx / WKLD_, d = idx - k * WKLD_;
  _Float16 v = (_Float16)0.f;
  if (d < ZT_) v = (_Float16)Wk[k * ZT_ + d];
  Wk16[idx] = v;
}

// ---------------- kernel 2: context-norm + tag ----------------
__global__ void norm_tag_kernel(const float* __restrict__ z,
                                const float* __restrict__ gamma,
                                const float* __restrict__ beta,
                                _Float16* __restrict__ zt) {
  int b = blockIdx.x;
  int d = threadIdx.x;                 // 0..127
  const float* zb = z + (size_t)b * T_ * Z_ + d;
  float s = 0.f, s2 = 0.f;
#pragma unroll 4
  for (int t = 0; t < T_; t++) { float v = zb[(size_t)t * Z_]; s += v; s2 += v * v; }
  float mu  = s * (1.f / T_);
  float var = (s2 - (float)T_ * mu * mu) * (1.f / (T_ - 1));   // ddof=1
  float inv = rsqrtf(var + 1e-8f);
  float g = gamma[d], be = beta[d];
  _Float16* ztb = zt + (size_t)b * FLAT_;
#pragma unroll 4
  for (int t = 0; t < T_; t++) {
    float v = (zb[(size_t)t * Z_] - mu) * inv * g + be;
    ztb[t * ZT_ + d] = (_Float16)v;
  }
  if (d == 0)
    for (int t = 0; t < T_; t++) ztb[t * ZT_ + Z_] = (_Float16)(float)t;
}

// ---------------- kernel 3: Q GEMM  Q[B,1024] = zt[B,16512] @ Wcat^T ----------
// Block: 8 waves, tile M=128 (16 rows/wave), N=64. B chunk (64 cols x 64 K)
// staged in LDS, double-buffered via async global->LDS copies.
__global__ __launch_bounds__(256) void qgemm_kernel(
    const _Float16* __restrict__ A,    // [B][16512]
    const _Float16* __restrict__ Wc,   // [1024][16512]  (BT layout)
    _Float16* __restrict__ Q) {        // [B][1024]
  __shared__ _Float16 sB[2 * 64 * SB_LD];          // 2 x 9216 B

  int tid = threadIdx.x;
  int wave = tid >> 5, lane = tid & 31;
  int lh = lane & 15, sel = lane >> 4;
  int m0 = blockIdx.y * 128 + wave * 16;
  int n0 = blockIdx.x * 64;

  // staging coords: 512 x 16B transfers per chunk, 2 per thread
  int u0 = tid * 2;
  int col0 = u0 >> 3, kp0 = (u0 & 7) * 8;
  int u1 = u0 + 1;
  int col1 = u1 >> 3, kp1 = (u1 & 7) * 8;
  const _Float16* gsrc0 = Wc + (size_t)(n0 + col0) * FLAT_ + kp0;
  const _Float16* gsrc1 = Wc + (size_t)(n0 + col1) * FLAT_ + kp1;
  _Float16* ldst0 = sB + col0 * SB_LD + kp0;
  _Float16* ldst1 = sB + col1 * SB_LD + kp1;
  const int BUF = 64 * SB_LD;

  const _Float16* arow = A + (size_t)(m0 + lh) * FLAT_ + sel * 8;

  v8f acc[4] = {v8f{}, v8f{}, v8f{}, v8f{}};

  // preamble: stage chunk 0 into buffer 0
#if HAS_ASYNC_LDS
  stage_async16(gsrc0, ldst0);
  stage_async16(gsrc1, ldst1);
  wait_async0();
#else
  { v8h t0 = *(const v8h*)gsrc0, t1 = *(const v8h*)gsrc1;
    *(v8h*)ldst0 = t0; *(v8h*)ldst1 = t1; }
#endif
  __syncthreads();

  const int NCHUNK = FLAT_ / QG_KC;                // 258
  for (int c = 0; c < NCHUNK; c++) {
    int cur = c & 1, nxt = cur ^ 1;
    bool more = (c + 1 < NCHUNK);
    int knext = (c + 1) * QG_KC;

    // kick off next chunk while computing on current one
#if HAS_ASYNC_LDS
    if (more) {
      stage_async16(gsrc0 + knext, ldst0 + nxt * BUF);
      stage_async16(gsrc1 + knext, ldst1 + nxt * BUF);
    }
#else
    v8h t0{}, t1{};
    if (more) { t0 = *(const v8h*)(gsrc0 + knext); t1 = *(const v8h*)(gsrc1 + knext); }
#endif

    int kbase = c * QG_KC;
    const _Float16* bbuf = sB + cur * BUF;
#pragma unroll
    for (int kk = 0; kk < 2; kk++) {
      int k0 = kbase + kk * 32;
      v16h a = combine16(*(const v8h*)(arow + k0), *(const v8h*)(arow + k0 + 16));
      // pre-load all 4 B fragments into distinct registers so the ds_loads
      // issue as a clause and overlap with the WMMAs (no per-frag dscnt stall)
      v16h bfr[4];
#pragma unroll
      for (int nt = 0; nt < 4; nt++)
        bfr[nt] = load_b_frag(bbuf, SB_LD, nt * 16, kk * 32, lane);
#pragma unroll
      for (int nt = 0; nt < 4; nt++)
        acc[nt] = wmma_f16(a, bfr[nt], acc[nt]);
    }

#if HAS_ASYNC_LDS
    if (more) wait_async0();
#else
    if (more) { *(v8h*)(ldst0 + nxt * BUF) = t0; *(v8h*)(ldst1 + nxt * BUF) = t1; }
#endif
    __syncthreads();
  }

#pragma unroll
  for (int nt = 0; nt < 4; nt++) {
#pragma unroll
    for (int i = 0; i < 8; i++) {
      int row = m0 + i + sel * 8;
      int col = n0 + nt * 16 + lh;
      Q[(size_t)row * NQ_ + col] = (_Float16)acc[nt][i];
    }
  }
}

// ---------------- kernel 4: per-batch attention + head ----------------
__global__ __launch_bounds__(256) void attend_kernel(
    const _Float16* __restrict__ zt,   // [B][16512]
    const _Float16* __restrict__ Qf,   // [B][1024]
    const _Float16* __restrict__ Wk16, // [16][160]
    const float* __restrict__ Ws,      // [16][129]
    const float* __restrict__ Wh,      // [8][576]
    const float* __restrict__ bh,      // [8]
    const float* __restrict__ Wy,      // [4][8]
    const float* __restrict__ by,      // [4]
    float* __restrict__ out) {
  __shared__ _Float16 s_zt[T_ * WKLD_];      // zt padded to K=160     (40960 B)
  __shared__ _Float16 s_ztT[WKLD_ * T_];     // transpose, rows d      (40960 B)
  __shared__ _Float16 s_q[64 * 32];          // heads x KS(pad32)      ( 4096 B)
  __shared__ _Float16 s_k[T_ * 32];          // keys, KS padded to 32  ( 8192 B)
  __shared__ float    s_s[T_ * 64];          // scores [t][head]       (32768 B)
  __shared__ _Float16 s_at[64 * T_];         // exp(A)^T [head][t]     (16384 B)
  __shared__ float    s_e[64 * 144];         // E [head][d(pad144)]    (36864 B)
  __shared__ float    s_inv[64];
  __shared__ float    s_R[NH_ * (NR_ + 2)];  // 576
  __shared__ float    s_h8[8];
  __shared__ float    s_y4[4];

  int b = blockIdx.x;
  int tid = threadIdx.x;
  int wave = tid >> 5, lane = tid & 31;
  int lh = lane & 15, sel = lane >> 4;

  const _Float16* ztb = zt + (size_t)b * FLAT_;

  // ---- stage zt (+transpose, zero-pad) ----
  for (int idx = tid; idx < T_ * WKLD_; idx += 256) {
    int t = idx / WKLD_, d = idx - t * WKLD_;
    _Float16 v = (d < ZT_) ? ztb[t * ZT_ + d] : (_Float16)0.f;
    s_zt[t * WKLD_ + d] = v;
    s_ztT[d * T_ + t]   = v;
  }
  // ---- stage Q heads (pad KS 16 -> 32 with zeros); head n col base = n*16 ----
  const _Float16* qb = Qf + (size_t)b * NQ_;
  for (int idx = tid; idx < 64 * 32; idx += 256) {
    int n = idx >> 5, k = idx & 31;
    _Float16 v = (_Float16)0.f;
    if (k < KS_) v = qb[n * KS_ + k];
    s_q[idx] = v;
  }
  // ---- zero key pad ----
  for (int idx = tid; idx < T_ * 32; idx += 256) s_k[idx] = (_Float16)0.f;
  __syncthreads();

  // ---- K = zt @ Wk^T  (8 M-tiles, K=160) ----
  {
    int m0 = wave * 16;
    v8f acc = {};
#pragma unroll
    for (int kk = 0; kk < 5; kk++) {
      v16h a = load_a_frag(s_zt, WKLD_, m0, kk * 32, lane);
      v16h bb = load_b_frag(Wk16, WKLD_, 0, kk * 32, lane);
      acc = wmma_f16(a, bb, acc);
    }
#pragma unroll
    for (int i = 0; i < 8; i++)
      s_k[(m0 + i + sel * 8) * 32 + lh] = (_Float16)acc[i];
  }
  __syncthreads();

  // ---- scores[t][head] = K @ Qcat^T  (8x4 tiles, K=32) ----
  {
    int m0 = wave * 16;
    v16h a = load_a_frag(s_k, 32, m0, 0, lane);
    v16h bfr[4];
#pragma unroll
    for (int nt = 0; nt < 4; nt++)
      bfr[nt] = load_b_frag(s_q, 32, nt * 16, 0, lane);
#pragma unroll
    for (int nt = 0; nt < 4; nt++) {
      v8f acc = {};
      acc = wmma_f16(a, bfr[nt], acc);
#pragma unroll
      for (int i = 0; i < 8; i++)
        s_s[(m0 + i + sel * 8) * 64 + nt * 16 + lh] = acc[i];
    }
  }
  __syncthreads();

  // ---- softmax over t per head column; store exp as f16 transposed ----
  if (tid < 64) {
    int h = tid;
    float mx = -1e30f;
    for (int t = 0; t < T_; t++) mx = fmaxf(mx, s_s[t * 64 + h]);
    float sum = 0.f;
    for (int t = 0; t < T_; t++) {
      float e = expf(s_s[t * 64 + h] - mx);
      sum += e;
      s_at[h * T_ + t] = (_Float16)e;
    }
    s_inv[h] = 1.f / sum;
  }
  __syncthreads();

  // ---- E = softmax(A)^T @ zt   (4x9 tiles, K=128) ----
  for (int tile = wave; tile < 36; tile += 8) {
    int mt = tile / 9, nt = tile - mt * 9;
    v8f acc = {};
#pragma unroll
    for (int kk = 0; kk < 4; kk++) {
      v16h a = load_a_frag(s_at, T_, mt * 16, kk * 32, lane);
      v16h bb = load_b_frag(s_ztT, T_, nt * 16, kk * 32, lane);
      acc = wmma_f16(a, bb, acc);
    }
#pragma unroll
    for (int i = 0; i < 8; i++) {
      int row = mt * 16 + i + sel * 8;
      s_e[row * 144 + nt * 16 + lh] = acc[i] * s_inv[row];
    }
  }
  __syncthreads();

  // ---- relations D and R vector ----
  for (int e = tid; e < NH_ * NR_; e += 256) {
    int h = e >> 4, r = e & 15;
    float acc = 0.f;
    for (int d = 0; d < ZT_; d++)
      acc += (s_e[h * 144 + d] - s_e[(h + NH_) * 144 + d]) * Ws[r * ZT_ + d];
    s_R[h * (NR_ + 2) + r] = acc;
  }
  if (tid < NH_) {
    s_R[tid * (NR_ + 2) + NR_]     = s_e[tid * 144 + Z_];
    s_R[tid * (NR_ + 2) + NR_ + 1] = s_e[(tid + NH_) * 144 + Z_];
  }
  __syncthreads();

  // ---- MLP head ----
  if (tid < 8) {
    float acc = bh[tid];
    for (int i = 0; i < NH_ * (NR_ + 2); i++) acc += s_R[i] * Wh[tid * NH_ * (NR_ + 2) + i];
    s_h8[tid] = fmaxf(acc, 0.f);
  }
  __syncthreads();
  if (tid < 4) {
    float acc = by[tid];
#pragma unroll
    for (int j = 0; j < 8; j++) acc += s_h8[j] * Wy[tid * 8 + j];
    s_y4[tid] = acc;
    out[(size_t)b * 4 + tid] = acc;
  }
  __syncthreads();
  if (tid == 0) {
    int best = 0; float bv = s_y4[0];
#pragma unroll
    for (int j = 1; j < 4; j++) if (s_y4[j] > bv) { bv = s_y4[j]; best = j; }
    out[(size_t)B_ * 4 + b] = (float)best;
  }
}

// ---------------- launch ----------------
extern "C" void kernel_launch(void* const* d_in, const int* in_sizes, int n_in,
                              void* d_out, int out_size, void* d_ws, size_t ws_size,
                              hipStream_t stream) {
  const float* z_seq = (const float*)d_in[0];
  const float* W_k   = (const float*)d_in[1];
  const float* W_q1  = (const float*)d_in[2];
  const float* W_q2  = (const float*)d_in[3];
  const float* W_s   = (const float*)d_in[4];
  const float* gamma = (const float*)d_in[5];
  const float* beta  = (const float*)d_in[6];
  const float* W_h   = (const float*)d_in[7];
  const float* b_h   = (const float*)d_in[8];
  const float* W_y   = (const float*)d_in[9];
  const float* b_y   = (const float*)d_in[10];
  float* out = (float*)d_out;

  char* ws = (char*)d_ws;
  _Float16* zt   = (_Float16*)ws; ws += (size_t)B_ * FLAT_ * sizeof(_Float16);
  _Float16* Wcat = (_Float16*)ws; ws += (size_t)NQ_ * FLAT_ * sizeof(_Float16);
  _Float16* Qf   = (_Float16*)ws; ws += (size_t)B_ * NQ_ * sizeof(_Float16);
  _Float16* Wk16 = (_Float16*)ws; ws += (size_t)KS_ * WKLD_ * sizeof(_Float16);

  {
    size_t total = (size_t)NQ_ * FLAT_;
    int blocks = (int)((total + 255) / 256);
    prep_wcat_kernel<<<blocks, 256, 0, stream>>>(W_q1, W_q2, Wcat);
  }
  prep_wk_kernel<<<(KS_ * WKLD_ + 255) / 256, 256, 0, stream>>>(W_k, Wk16);
  norm_tag_kernel<<<B_, 128, 0, stream>>>(z_seq, gamma, beta, zt);
  qgemm_kernel<<<dim3(NQ_ / 64, B_ / 128), 256, 0, stream>>>(zt, Wcat, Qf);
  attend_kernel<<<B_, 256, 0, stream>>>(zt, Qf, Wk16, W_s, W_h, b_h, W_y, b_y, out);
}
